// GAT_Model_77653008712213
// MI455X (gfx1250) — matrix operand order repaired and checked by reading, not compiled, not run on hardware
//
#include <hip/hip_runtime.h>
#include <hip/hip_bf16.h>

// ---------------------------------------------------------------------------
// Types for CDNA5 WMMA (wave32): v_wmma_f32_16x16x32_bf16
// ---------------------------------------------------------------------------
typedef __bf16 bf16x16 __attribute__((ext_vector_type(16)));
typedef float  f32x8   __attribute__((ext_vector_type(8)));

__device__ __forceinline__ unsigned short f2bf(float f) {
    unsigned u = __float_as_uint(f);
    unsigned r = u + 0x7FFFu + ((u >> 16) & 1u);   // round-to-nearest-even
    return (unsigned short)(r >> 16);
}
// ordered-uint encoding so unsigned atomicMax == float max
__device__ __forceinline__ unsigned f2o(float f) {
    unsigned u = __float_as_uint(f);
    return (u & 0x80000000u) ? ~u : (u | 0x80000000u);
}
__device__ __forceinline__ float o2f(unsigned u) {
    return (u & 0x80000000u) ? __uint_as_float(u & 0x7FFFFFFFu) : __uint_as_float(~u);
}

// ---------------------------------------------------------------------------
// Zero fill
// ---------------------------------------------------------------------------
__global__ void k_fill0(unsigned* __restrict__ p, int n) {
    int t = blockIdx.x * blockDim.x + threadIdx.x;
    if (t < n) p[t] = 0u;
}

// ---------------------------------------------------------------------------
// Pack f32 [Mused x srcStride] -> bf16 A tiles in WMMA 16x32 per-lane layout.
// dst index: ((mt*ktiles + kt)*32 + lane)*16 + i
// lane<16: elems 0..7 -> K=kt*32+0..7,  elems 8..15 -> K=kt*32+16..23
// lane>=16: shift K base by +8.   row M = mt*16 + (lane&15)
// ---------------------------------------------------------------------------
__global__ void k_pack_a(const float* __restrict__ src, int srcStride,
                         int Mused, int Kused, int ktiles,
                         unsigned short* __restrict__ dst, int total) {
    int t = blockIdx.x * blockDim.x + threadIdx.x;
    if (t >= total) return;
    int i    = t & 15;
    int lane = (t >> 4) & 31;
    int tile = t >> 9;
    int kt = tile % ktiles;
    int mt = tile / ktiles;
    int m = mt * 16 + (lane & 15);
    int k = kt * 32 + (i < 8 ? i : i + 8) + ((lane & 16) >> 1);
    float v = (m < Mused && k < Kused) ? src[(size_t)m * srcStride + k] : 0.f;
    dst[t] = f2bf(v);
}

// ---------------------------------------------------------------------------
// Pack f32 weight [Kused x Nused] row-major -> bf16 B tiles (32x16 per-lane).
// dst index: ((kt*ntiles + nt)*32 + lane)*16 + i
// lane<16: elems 0..7 -> K=kt*32+0..7, elems 8..15 -> K=kt*32+16..23 ; lane>=16: +8 ; col N = nt*16+(lane&15)
// ---------------------------------------------------------------------------
__global__ void k_pack_b(const float* __restrict__ W, int Kused, int Nused,
                         int ntiles, unsigned short* __restrict__ dst, int total) {
    int t = blockIdx.x * blockDim.x + threadIdx.x;
    if (t >= total) return;
    int i    = t & 15;
    int lane = (t >> 4) & 31;
    int tile = t >> 9;
    int nt = tile % ntiles;
    int kt = tile / ntiles;
    int n = nt * 16 + (lane & 15);
    int k = kt * 32 + (i < 8 ? i : i + 8) + ((lane & 16) >> 1);
    float v = (k < Kused && n < Nused) ? W[(size_t)k * Nused + n] : 0.f;
    dst[t] = f2bf(v);
}

// ---------------------------------------------------------------------------
// WMMA GEMM: C[Mpad x Npad] (f32) = A(bf16 packed) * B(bf16 packed) + bias
// 8 waves / block; each wave owns one 16x16 output tile.
// ---------------------------------------------------------------------------
__global__ void k_wmma_gemm(const unsigned short* __restrict__ Apk,
                            const unsigned short* __restrict__ Bpk,
                            const float* __restrict__ bias, int Nused,
                            float* __restrict__ C,
                            int Kpad, int Npad, int relu) {
    const int lane   = threadIdx.x & 31;
    const int wave   = threadIdx.x >> 5;
    const int ntiles = Npad >> 4;
    const int ktiles = Kpad >> 5;
    const int nt = blockIdx.x * 8 + wave;
    const int mt = blockIdx.y;
    if (nt >= ntiles) return;

    const unsigned short* aptr = Apk + ((size_t)mt * ktiles * 32 + lane) * 16;
    const unsigned short* bptr = Bpk + ((size_t)nt * 32 + lane) * 16;
    const size_t bstride = (size_t)ntiles * 512;

    f32x8 acc = {};
    for (int kt = 0; kt < ktiles; ++kt) {
        __builtin_prefetch((const void*)(bptr + bstride), 0, 1);
        bf16x16 a = *(const bf16x16*)aptr;
        bf16x16 b = *(const bf16x16*)bptr;
        acc = __builtin_amdgcn_wmma_f32_16x16x32_bf16(
            /*neg_a=*/false, a, /*neg_b=*/false, b,
            /*c_mod=*/(short)0, acc, /*reuse_a=*/false, /*reuse_b=*/false);
        aptr += 512;
        bptr += bstride;
    }

    const int col   = nt * 16 + (lane & 15);
    const int rbase = mt * 16 + ((lane & 16) >> 1);
    const float bv  = (col < Nused) ? bias[col] : 0.f;
#pragma unroll
    for (int v = 0; v < 8; ++v) {
        float val = acc[v] + bv;
        if (relu) val = fmaxf(val, 0.f);
        C[(size_t)(rbase + v) * Npad + col] = val;
    }
}

// ---------------------------------------------------------------------------
// Attention logits: als[n,h] = sum_c h[n,h*C+c]*a_src[h,c]; ald likewise.
// ---------------------------------------------------------------------------
__global__ void k_attn_logits(const float* __restrict__ hf, int Np, int H, int C, int N,
                              const float* __restrict__ asrc, const float* __restrict__ adst,
                              float* __restrict__ als, float* __restrict__ ald) {
    int t = blockIdx.x * blockDim.x + threadIdx.x;
    if (t >= N * H) return;
    int n = t / H, h = t - n * H;
    const float* row = hf + (size_t)n * Np + h * C;
    const float* a1 = asrc + h * C;
    const float* a2 = adst + h * C;
    float s1 = 0.f, s2 = 0.f;
    for (int c = 0; c < C; ++c) { float x = row[c]; s1 += x * a1[c]; s2 += x * a2[c]; }
    als[t] = s1; ald[t] = s2;
}

// ---------------------------------------------------------------------------
// Edge pass 1: segment max of leaky_relu(als[src]+ald[dst]) over dst
// ---------------------------------------------------------------------------
__global__ void k_edge_max(const int* __restrict__ ei, int E, int N, int H,
                           const float* __restrict__ als, const float* __restrict__ ald,
                           unsigned* __restrict__ menc) {
    int t = blockIdx.x * blockDim.x + threadIdx.x;
    int total = (E + N) * H;
    if (t >= total) return;
    int e = t / H, h = t - e * H;
    int s, d;
    if (e < E) { s = ei[e]; d = ei[E + e]; } else { s = e - E; d = s; }
    float v = als[s * H + h] + ald[d * H + h];
    v = v > 0.f ? v : 0.2f * v;
    atomicMax(&menc[d * H + h], f2o(v));
}

// ---------------------------------------------------------------------------
// Edge pass 2: exp(e - max); store per-edge value; segment sum over dst
// ---------------------------------------------------------------------------
__global__ void k_edge_sum(const int* __restrict__ ei, int E, int N, int H,
                           const float* __restrict__ als, const float* __restrict__ ald,
                           const unsigned* __restrict__ menc,
                           float* __restrict__ expv, float* __restrict__ ssum) {
    int t = blockIdx.x * blockDim.x + threadIdx.x;
    int total = (E + N) * H;
    if (t >= total) return;
    int e = t / H, h = t - e * H;
    int s, d;
    if (e < E) { s = ei[e]; d = ei[E + e]; } else { s = e - E; d = s; }
    float v = als[s * H + h] + ald[d * H + h];
    v = v > 0.f ? v : 0.2f * v;
    float ex = __expf(v - o2f(menc[d * H + h]));
    expv[t] = ex;
    atomicAdd(&ssum[d * H + h], ex);
}

// ---------------------------------------------------------------------------
// Edge pass 3: agg[dst, h*C+c] += h[src, h*C+c] * alpha[e,h]
// ---------------------------------------------------------------------------
__global__ void k_edge_scatter(const int* __restrict__ ei, int E, int N, int H, int C, int Np,
                               const float* __restrict__ hf,
                               const float* __restrict__ expv, const float* __restrict__ ssum,
                               float* __restrict__ agg) {
    int t = blockIdx.x * blockDim.x + threadIdx.x;
    int HC = H * C;
    int total = (E + N) * HC;
    if (t >= total) return;
    int e = t / HC, r = t - e * HC;
    int h = r / C;
    int s, d;
    if (e < E) { s = ei[e]; d = ei[E + e]; } else { s = e - E; d = s; }
    float alpha = expv[e * H + h] / (ssum[d * H + h] + 1e-16f);
    atomicAdd(&agg[(size_t)d * Np + r], hf[(size_t)s * Np + r] * alpha);
}

// ---------------------------------------------------------------------------
// Finalize conv layer: agg = relu(agg + bias)   (in place, used cols only)
// ---------------------------------------------------------------------------
__global__ void k_finalize(float* __restrict__ agg, const float* __restrict__ b,
                           int N, int HC, int Np) {
    int t = blockIdx.x * blockDim.x + threadIdx.x;
    if (t >= N * HC) return;
    int n = t / HC, r = t - n * HC;
    float v = agg[(size_t)n * Np + r] + b[r];
    agg[(size_t)n * Np + r] = fmaxf(v, 0.f);
}

// ---------------------------------------------------------------------------
// global_add_pool: pooled[batch[n], c] += h[n, c]
// ---------------------------------------------------------------------------
__global__ void k_pool(const float* __restrict__ hf, const int* __restrict__ batch,
                       int N, int Cu, int Np, int Pp, float* __restrict__ pooled) {
    int t = blockIdx.x * blockDim.x + threadIdx.x;
    if (t >= N * Cu) return;
    int n = t / Cu, c = t - n * Cu;
    atomicAdd(&pooled[(size_t)batch[n] * Pp + c], hf[(size_t)n * Np + c]);
}

// ---------------------------------------------------------------------------
// z = concat(g[B,1024], desc[B,1217]) into f32 [B, 2272] (zero padded)
// ---------------------------------------------------------------------------
__global__ void k_concat(const float* __restrict__ g, const float* __restrict__ desc,
                         float* __restrict__ z, int B, int G, int D, int Zp) {
    int t = blockIdx.x * blockDim.x + threadIdx.x;
    if (t >= B * Zp) return;
    int b = t / Zp, j = t - b * Zp;
    float v = 0.f;
    if (j < G)            v = g[(size_t)b * G + j];
    else if (j < G + D)   v = desc[(size_t)b * D + (j - G)];
    z[t] = v;
}

__global__ void k_copy_out(const float* __restrict__ src, int stride, float* __restrict__ out, int B) {
    int t = blockIdx.x * blockDim.x + threadIdx.x;
    if (t < B) out[t] = src[(size_t)t * stride];
}

// ---------------------------------------------------------------------------
// Host driver
// ---------------------------------------------------------------------------
static inline int cdiv(int a, int b) { return (a + b - 1) / b; }

extern "C" void kernel_launch(void* const* d_in, const int* in_sizes, int n_in,
                              void* d_out, int out_size, void* d_ws, size_t ws_size,
                              hipStream_t stream) {
    (void)in_sizes; (void)n_in; (void)out_size; (void)ws_size;
    const int N = 10000, E = 160000, B = 256, DESC = 1217;
    const int Et = E + N;

    const float* x      = (const float*)d_in[0];
    const int*   ei     = (const int*)  d_in[1];
    const int*   batch  = (const int*)  d_in[2];
    const float* desc   = (const float*)d_in[3];
    float* out = (float*)d_out;

    // --- workspace bump allocator (256B aligned) ---
    size_t off = 0;
    auto alloc = [&](size_t bytes) -> void* {
        void* p = (char*)d_ws + off;
        off += (bytes + 255) & ~(size_t)255;
        return p;
    };
    float*          hf     = (float*)         alloc((size_t)N * 864 * 4);  // gemm out / fc ping
    float*          agg    = (float*)         alloc((size_t)N * 864 * 4);  // aggregate / fc pong
    unsigned short* apk    = (unsigned short*)alloc((size_t)N * 864 * 2);  // packed A
    unsigned short* bpk    = (unsigned short*)alloc((size_t)2272 * 2048 * 2); // packed B
    float*          als    = (float*)         alloc((size_t)N * 5 * 4);
    float*          ald    = (float*)         alloc((size_t)N * 5 * 4);
    unsigned*       menc   = (unsigned*)      alloc((size_t)N * 5 * 4);
    float*          ssum   = (float*)         alloc((size_t)N * 5 * 4);
    float*          expv   = (float*)         alloc((size_t)Et * 5 * 4);
    float*          pooled = (float*)         alloc((size_t)B * 576 * 4);

    const int TB = 256;

    // GEMM helper (host lambda): pack B, pack A, run WMMA gemm.
    auto gemm = [&](const float* srcA, int strideA, int Mused, int Mpad,
                    const float* W, int Kused, int Kpad, int Nused, int Npad,
                    const float* bias, int relu, float* Cout) {
        int ktiles = Kpad >> 5, ntiles = Npad >> 4, mtiles = Mpad >> 4;
        int totB = ktiles * ntiles * 512;
        int totA = mtiles * ktiles * 512;
        k_pack_b<<<cdiv(totB, TB), TB, 0, stream>>>(W, Kused, Nused, ntiles, bpk, totB);
        k_pack_a<<<cdiv(totA, TB), TB, 0, stream>>>(srcA, strideA, Mused, Kused, ktiles, apk, totA);
        dim3 grid(cdiv(ntiles, 8), mtiles);
        k_wmma_gemm<<<grid, 256, 0, stream>>>(apk, bpk, bias, Nused, Cout, Kpad, Npad, relu);
    };

    // --- GAT conv layers ---
    struct ConvP { int wi; int K, Kp, H, C, Np; };
    const ConvP cp[4] = {
        {4,  57,  64,  5, 114, 576},
        {8,  570, 576, 5, 171, 864},
        {12, 855, 864, 5, 114, 576},
        {16, 570, 576, 1, 570, 576},
    };
    const float* srcA = x;
    int strideA = 57;
    int prevNp = 57;
    for (int L = 0; L < 4; ++L) {
        const ConvP& p = cp[L];
        const float* W  = (const float*)d_in[p.wi + 0];
        const float* as = (const float*)d_in[p.wi + 1];
        const float* ad = (const float*)d_in[p.wi + 2];
        const float* bb = (const float*)d_in[p.wi + 3];
        const int HC = p.H * p.C;

        // h = x @ W   (bias added in finalize along with aggregation)
        gemm(srcA, strideA, N, N, W, p.K, p.Kp, HC, p.Np, bb /*zeros; harmless*/, 0, hf);

        // attention logits
        k_attn_logits<<<cdiv(N * p.H, TB), TB, 0, stream>>>(hf, p.Np, p.H, p.C, N, as, ad, als, ald);

        // segment softmax + scatter
        k_fill0<<<cdiv(N * p.H, TB), TB, 0, stream>>>(menc, N * p.H);
        k_fill0<<<cdiv(N * p.H, TB), TB, 0, stream>>>((unsigned*)ssum, N * p.H);
        k_edge_max<<<cdiv(Et * p.H, TB), TB, 0, stream>>>(ei, E, N, p.H, als, ald, menc);
        k_edge_sum<<<cdiv(Et * p.H, TB), TB, 0, stream>>>(ei, E, N, p.H, als, ald, menc, expv, ssum);
        k_fill0<<<cdiv(N * p.Np, TB), TB, 0, stream>>>((unsigned*)agg, N * p.Np);
        k_edge_scatter<<<cdiv(Et * HC, TB), TB, 0, stream>>>(ei, E, N, p.H, p.C, p.Np, hf, expv, ssum, agg);

        // out = relu(agg + b)
        k_finalize<<<cdiv(N * HC, TB), TB, 0, stream>>>(agg, bb, N, HC, p.Np);

        srcA = agg; strideA = p.Np; prevNp = p.Np;
    }

    // --- global add pool -> [B, 570] (stride 576) ---
    k_fill0<<<cdiv(B * 576, TB), TB, 0, stream>>>((unsigned*)pooled, B * 576);
    k_pool<<<cdiv(N * 570, TB), TB, 0, stream>>>(agg, batch, N, 570, prevNp, 576, pooled);

    // --- g = pooled @ Wg + bg ---  -> hf [B,1024]
    gemm(pooled, 576, B, B, (const float*)d_in[20], 570, 576, 1024, 1024,
         (const float*)d_in[21], 0, hf);

    // --- z = concat(g, desc) -> agg [B, 2272] ---
    k_concat<<<cdiv(B * 2272, TB), TB, 0, stream>>>(hf, desc, agg, B, 1024, DESC, 2272);

    // --- FC stack ---
    gemm(agg, 2272, B, B, (const float*)d_in[22], 2241, 2272, 1024, 1024,
         (const float*)d_in[23], 1, hf);                                   // FC1
    gemm(hf, 1024, B, B, (const float*)d_in[24], 1024, 1024, 2048, 2048,
         (const float*)d_in[25], 1, agg);                                  // FC2
    gemm(agg, 2048, B, B, (const float*)d_in[26], 2048, 2048, 1024, 1024,
         (const float*)d_in[27], 1, hf);                                   // FC3
    gemm(hf, 1024, B, B, (const float*)d_in[28], 1024, 1024, 256, 256,
         (const float*)d_in[29], 1, agg);                                  // FC4
    gemm(agg, 256, B, B, (const float*)d_in[30], 256, 256, 1, 16,
         (const float*)d_in[31], 0, hf);                                   // FC5 -> [B,16]

    k_copy_out<<<cdiv(B, TB), TB, 0, stream>>>(hf, 16, out, B);
}
